// Snet_74277164417394
// MI455X (gfx1250) — matrix operand order repaired and checked by reading, not compiled, not run on hardware
//
#include <hip/hip_runtime.h>

// ---------------------------------------------------------------------------
// Problem constants (match reference)
// ---------------------------------------------------------------------------
constexpr int BATCH = 2048;
constexpr int CIN   = 256;
constexpr int LIN   = 256;
constexpr int COUT  = 64;
constexpr int LOUT  = 61;            // (256-16)/4 + 1
constexpr int FLATSZ = COUT * LOUT;  // 3904
constexpr int H1    = 512;
constexpr int H2    = 128;
constexpr int KCONV = CIN * 16;      // 4096

// ---------------------------------------------------------------------------
// Vector types for WMMA
// ---------------------------------------------------------------------------
typedef __attribute__((ext_vector_type(16))) __bf16 v16bf;
typedef __attribute__((ext_vector_type(8)))  __bf16 v8bf;
typedef __attribute__((ext_vector_type(4)))  __bf16 v4bf;
typedef __attribute__((ext_vector_type(8)))  float  v8f;
typedef __attribute__((ext_vector_type(4)))  float  v4f;

static __device__ __forceinline__ v8f wmma_bf16(v16bf a, v16bf b, v8f c) {
  // D = A(16x32 bf16) * B(32x16 bf16) + C(16x16 f32)
  return __builtin_amdgcn_wmma_f32_16x16x32_bf16(
      /*neg_a=*/false, a, /*neg_b=*/false, b,
      /*c_mod=*/(short)0, c, /*reuse_a=*/false, /*reuse_b=*/false);
}

// ---------------------------------------------------------------------------
// Prep kernels
// ---------------------------------------------------------------------------
__global__ void cvt_f32_to_bf16(const float* __restrict__ src,
                                __bf16* __restrict__ dst, int n) {
  int i = blockIdx.x * 256 + threadIdx.x;
  if (i < n) dst[i] = (__bf16)src[i];
}

__global__ void bn_fold_kernel(const float* __restrict__ gamma,
                               const float* __restrict__ beta,
                               const float* __restrict__ mean,
                               const float* __restrict__ var,
                               const float* __restrict__ conv_b,
                               float* __restrict__ scale,
                               float* __restrict__ shift) {
  int c = threadIdx.x;
  if (c < COUT) {
    float inv = gamma[c] * rsqrtf(var[c] + 1e-5f);
    scale[c] = inv;
    shift[c] = (conv_b[c] - mean[c]) * inv + beta[c];
  }
}

// ---------------------------------------------------------------------------
// Conv1d + BN + ReLU as implicit-im2col GEMM.
// grid = BATCH blocks of 128 threads (4 waves). Wave w: l-rows [16w, 16w+15].
// K-chunk of 32 = input channels {2j, 2j+1} x 16 taps.
// A path: x[b] streamed through LDS (bf16, double-buffered, 4 chunks = 8
// channels per staging group) so HBM reads x exactly once; all 4 waves share
// the staged rows and read their overlapping im2col windows via ds_load.
// B path: 4 N-tiles software-pipelined one K-chunk ahead in registers.
// ---------------------------------------------------------------------------
__global__ __launch_bounds__(128) void conv_bn_relu_kernel(
    const float*  __restrict__ x,      // [B, 256, 256]
    const __bf16* __restrict__ Wc,     // [64, 4096] bf16
    const float*  __restrict__ scale,  // [64]
    const float*  __restrict__ shift,  // [64]
    __bf16*       __restrict__ act)    // [B, 3904]  (c*61 + l)
{
  __shared__ __bf16 smem[2][8 * LIN];  // 2 x 4KB: 8 channel-rows of bf16

  const int b    = blockIdx.x;
  const int tid  = threadIdx.x;
  const int lane = tid & 31;
  const int wave = tid >> 5;      // 0..3
  const int lr   = lane & 15;     // row/col within 16
  const int kh   = lane >> 4;     // K-half select (contiguous k = 16*kh + s in A and B)

  const int row = wave * 16 + lr;                 // logical l index 0..63
  const int l   = row < LOUT ? row : (LOUT - 1);  // clamp pad rows

  // Staging role: thread t handles 16 contiguous elements of channel t>>4.
  const int scc  = tid >> 4;          // 0..7 channel within group
  const int spos = (tid & 15) * 16;   // 0..240
  const float* sg_base = x + (size_t)b * CIN * LIN + (size_t)scc * LIN + spos;

  // B: lane needs 16 contiguous bf16 of Wc row n = t*16+lr at k = chunk*32+kh*16
  const __bf16* wb = Wc + (size_t)lr * KCONV + kh * 16;

  v8f acc[4] = {v8f{}, v8f{}, v8f{}, v8f{}};

  // ---- preload B tiles for chunk 0 (pipelined one chunk ahead) ----
  v16bf b0 = *(const v16bf*)(wb);
  v16bf b1 = *(const v16bf*)(wb + 16 * KCONV);
  v16bf b2 = *(const v16bf*)(wb + 32 * KCONV);
  v16bf b3 = *(const v16bf*)(wb + 48 * KCONV);

  // ---- preload staging group 0 ----
  {
    const float* p = sg_base;
    v4f r0 = *(const v4f*)(p);
    v4f r1 = *(const v4f*)(p + 4);
    v4f r2 = *(const v4f*)(p + 8);
    v4f r3 = *(const v4f*)(p + 12);
    v8f q0 = __builtin_shufflevector(r0, r1, 0, 1, 2, 3, 4, 5, 6, 7);
    v8f q1 = __builtin_shufflevector(r2, r3, 0, 1, 2, 3, 4, 5, 6, 7);
    v8bf h0 = __builtin_convertvector(q0, v8bf);
    v8bf h1 = __builtin_convertvector(q1, v8bf);
    *(v8bf*)(&smem[0][scc * LIN + spos])     = h0;
    *(v8bf*)(&smem[0][scc * LIN + spos + 8]) = h1;
  }
  __syncthreads();

  for (int g = 0; g < 32; ++g) {              // 32 groups x 4 chunks = 128
    // Issue global loads for the NEXT staging group before compute.
    v4f r0{}, r1{}, r2{}, r3{};
    const bool havenext = (g + 1) < 32;
    const float* pnext = sg_base + (size_t)(g + 1) * 8 * LIN;
    if (havenext) {
      r0 = *(const v4f*)(pnext);
      r1 = *(const v4f*)(pnext + 4);
      r2 = *(const v4f*)(pnext + 8);
      r3 = *(const v4f*)(pnext + 12);
      if ((g + 2) < 32) __builtin_prefetch(pnext + 8 * LIN, 0, 3);
    }

    const __bf16* sb = smem[g & 1];
#pragma unroll
    for (int j = 0; j < 4; ++j) {
      // Issue next chunk's B tiles BEFORE consuming the current ones, so the
      // loads retire a full chunk of WMMA work later (clamped at the end).
      const int nextc = g * 4 + j + 1;
      const __bf16* pbn = wb + (size_t)(nextc < 128 ? nextc : 127) * 32;
      v16bf n0 = *(const v16bf*)(pbn);
      v16bf n1 = *(const v16bf*)(pbn + 16 * KCONV);
      v16bf n2 = *(const v16bf*)(pbn + 32 * KCONV);
      v16bf n3 = *(const v16bf*)(pbn + 48 * KCONV);

      // A from LDS: rows 2j (K 0..15) and 2j+1 (K 16..31) of this group.
      const __bf16* arow = sb + (2 * j + kh) * LIN + l * 4;
      v4bf a00 = *(const v4bf*)(arow);
      v4bf a01 = *(const v4bf*)(arow + 4);
      v4bf a10 = *(const v4bf*)(arow + 8);
      v4bf a11 = *(const v4bf*)(arow + 12);
      v8bf h0 = __builtin_shufflevector(a00, a01, 0, 1, 2, 3, 4, 5, 6, 7);
      v8bf h1 = __builtin_shufflevector(a10, a11, 0, 1, 2, 3, 4, 5, 6, 7);
      v16bf a = __builtin_shufflevector(h0, h1, 0, 1, 2, 3, 4, 5, 6, 7,
                                        8, 9, 10, 11, 12, 13, 14, 15);

      acc[0] = wmma_bf16(a, b0, acc[0]);
      acc[1] = wmma_bf16(a, b1, acc[1]);
      acc[2] = wmma_bf16(a, b2, acc[2]);
      acc[3] = wmma_bf16(a, b3, acc[3]);

      b0 = n0; b1 = n1; b2 = n2; b3 = n3;
    }

    if (havenext) {
      v8f q0 = __builtin_shufflevector(r0, r1, 0, 1, 2, 3, 4, 5, 6, 7);
      v8f q1 = __builtin_shufflevector(r2, r3, 0, 1, 2, 3, 4, 5, 6, 7);
      v8bf h0 = __builtin_convertvector(q0, v8bf);
      v8bf h1 = __builtin_convertvector(q1, v8bf);
      __bf16* sn = smem[(g + 1) & 1];
      *(v8bf*)(&sn[scc * LIN + spos])     = h0;
      *(v8bf*)(&sn[scc * LIN + spos + 8]) = h1;
    }
    __syncthreads();
  }

  // Epilogue: scale/shift (folded conv bias & BN), ReLU, store bf16.
  // D layout: VGPR r -> M = r + kh*8 ; N = lane&15.
  __bf16* actb = act + (size_t)b * FLATSZ;
#pragma unroll
  for (int t = 0; t < 4; ++t) {
    const int n = t * 16 + lr;
    const float s  = scale[n];
    const float sh = shift[n];
#pragma unroll
    for (int r = 0; r < 8; ++r) {
      const int lidx = wave * 16 + kh * 8 + r;
      if (lidx < LOUT) {
        float v = acc[t][r] * s + sh;
        v = v > 0.0f ? v : 0.0f;
        actb[n * LOUT + lidx] = (__bf16)v;
      }
    }
  }
}

// ---------------------------------------------------------------------------
// FC1: [2048,3904] x [3904,512] + bias -> h1 bf16 [2048,512]
// grid = 128 Mtiles * 4 Ngroups; block 256 = 8 waves, wave w -> ntile ng*8+w.
// One-deep software pipeline on A/B loads.
// ---------------------------------------------------------------------------
__global__ __launch_bounds__(256) void fc1_kernel(
    const __bf16* __restrict__ act,   // [2048, 3904]
    const __bf16* __restrict__ Wf,    // [512, 3904]
    const float*  __restrict__ bias,  // [512]
    __bf16*       __restrict__ h1)    // [2048, 512]
{
  const int mt   = blockIdx.x >> 2;
  const int ng   = blockIdx.x & 3;
  const int tid  = threadIdx.x;
  const int lane = tid & 31;
  const int wave = tid >> 5;
  const int lr   = lane & 15;
  const int kh   = lane >> 4;

  const int n = (ng * 8 + wave) * 16 + lr;   // 0..511
  const int m = mt * 16 + lr;                // 0..2047

  const __bf16* pa_base = act + (size_t)m * FLATSZ + kh * 16;
  const __bf16* pb_base = Wf  + (size_t)n * FLATSZ + kh * 16;

  auto loadA = [&](int chunk) -> v16bf {
    const __bf16* pa = pa_base + chunk * 32;
    v8bf a0 = *(const v8bf*)(pa);
    v8bf a1 = *(const v8bf*)(pa + 8);
    return __builtin_shufflevector(a0, a1, 0, 1, 2, 3, 4, 5, 6, 7,
                                   8, 9, 10, 11, 12, 13, 14, 15);
  };
  auto loadB = [&](int chunk) -> v16bf {
    return *(const v16bf*)(pb_base + chunk * 32);
  };

  v8f acc = {};
  v16bf aC = loadA(0);
  v16bf bC = loadB(0);
  constexpr int NCHUNK = FLATSZ / 32;  // 122
  for (int chunk = 0; chunk < NCHUNK - 1; ++chunk) {
    v16bf aN = loadA(chunk + 1);
    v16bf bN = loadB(chunk + 1);
    acc = wmma_bf16(aC, bC, acc);
    aC = aN; bC = bN;
  }
  acc = wmma_bf16(aC, bC, acc);

  const float bb = bias[n];
#pragma unroll
  for (int r = 0; r < 8; ++r) {
    const int mm = mt * 16 + kh * 8 + r;
    h1[(size_t)mm * H1 + n] = (__bf16)(acc[r] + bb);
  }
}

// ---------------------------------------------------------------------------
// FC2 + parabit heads: feats = h1 x W2^T + b2 ; out[b,i,c] = feats*bit_w+bit_b
// grid = 128 Mtiles; block 256 = 8 waves; wave w -> ntile w (N=128).
// ---------------------------------------------------------------------------
__global__ __launch_bounds__(256) void fc2_parabit_kernel(
    const __bf16* __restrict__ h1,     // [2048, 512]
    const __bf16* __restrict__ Wf2,    // [128, 512]
    const float*  __restrict__ bias2,  // [128]
    const float*  __restrict__ bit_w,  // [128, 2]
    const float*  __restrict__ bit_b,  // [128, 2]
    float*        __restrict__ out)    // [2048, 128, 2]
{
  const int mt   = blockIdx.x;
  const int tid  = threadIdx.x;
  const int lane = tid & 31;
  const int wave = tid >> 5;
  const int lr   = lane & 15;
  const int kh   = lane >> 4;

  const int n = wave * 16 + lr;   // head index i, 0..127
  const int m = mt * 16 + lr;

  const __bf16* pa_base = h1  + (size_t)m * H1 + kh * 16;
  const __bf16* pb_base = Wf2 + (size_t)n * H1 + kh * 16;

  auto loadA = [&](int chunk) -> v16bf {
    const __bf16* pa = pa_base + chunk * 32;
    v8bf a0 = *(const v8bf*)(pa);
    v8bf a1 = *(const v8bf*)(pa + 8);
    return __builtin_shufflevector(a0, a1, 0, 1, 2, 3, 4, 5, 6, 7,
                                   8, 9, 10, 11, 12, 13, 14, 15);
  };
  auto loadB = [&](int chunk) -> v16bf {
    return *(const v16bf*)(pb_base + chunk * 32);
  };

  v8f acc = {};
  v16bf aC = loadA(0);
  v16bf bC = loadB(0);
  constexpr int NCHUNK = H1 / 32;  // 16
#pragma unroll 4
  for (int chunk = 0; chunk < NCHUNK - 1; ++chunk) {
    v16bf aN = loadA(chunk + 1);
    v16bf bN = loadB(chunk + 1);
    acc = wmma_bf16(aC, bC, acc);
    aC = aN; bC = bN;
  }
  acc = wmma_bf16(aC, bC, acc);

  const float fb = bias2[n];
  const float w0 = bit_w[2 * n], w1 = bit_w[2 * n + 1];
  const float b0 = bit_b[2 * n], b1 = bit_b[2 * n + 1];
#pragma unroll
  for (int r = 0; r < 8; ++r) {
    const int mm = mt * 16 + kh * 8 + r;
    const float f = acc[r] + fb;
    float2 o = make_float2(f * w0 + b0, f * w1 + b1);
    *(float2*)(out + ((size_t)mm * H2 + n) * 2) = o;
  }
}

// ---------------------------------------------------------------------------
// Launcher
// ---------------------------------------------------------------------------
extern "C" void kernel_launch(void* const* d_in, const int* in_sizes, int n_in,
                              void* d_out, int out_size, void* d_ws, size_t ws_size,
                              hipStream_t stream) {
  const float* x      = (const float*)d_in[0];
  const float* conv_w = (const float*)d_in[1];
  const float* conv_b = (const float*)d_in[2];
  const float* bn_g   = (const float*)d_in[3];
  const float* bn_be  = (const float*)d_in[4];
  const float* bn_mu  = (const float*)d_in[5];
  const float* bn_var = (const float*)d_in[6];
  const float* fc1_w  = (const float*)d_in[7];
  const float* fc1_b  = (const float*)d_in[8];
  const float* fc2_w  = (const float*)d_in[9];
  const float* fc2_b  = (const float*)d_in[10];
  const float* bit_w  = (const float*)d_in[11];
  const float* bit_b  = (const float*)d_in[12];
  float* out = (float*)d_out;

  // Workspace layout (all offsets 256-B aligned), ~22.7 MB total
  char* ws = (char*)d_ws;
  __bf16* Wc    = (__bf16*)(ws + 0);         //   524288 B : conv_w bf16 [64,4096]
  float*  scale = (float*)(ws + 524288);     //      256 B
  float*  shift = (float*)(ws + 524544);     //      256 B
  __bf16* Wf1   = (__bf16*)(ws + 524800);    //  3997696 B : fc1_w bf16 [512,3904]
  __bf16* Wf2   = (__bf16*)(ws + 4522496);   //   131072 B : fc2_w bf16 [128,512]
  __bf16* act   = (__bf16*)(ws + 4653568);   // 15990784 B : act bf16 [2048,3904]
  __bf16* h1    = (__bf16*)(ws + 20644352);  //  2097152 B : h1 bf16 [2048,512]

  // Prep: weight conversions + BN fold
  cvt_f32_to_bf16<<<(COUT * KCONV + 255) / 256, 256, 0, stream>>>(conv_w, Wc, COUT * KCONV);
  cvt_f32_to_bf16<<<(H1 * FLATSZ + 255) / 256, 256, 0, stream>>>(fc1_w, Wf1, H1 * FLATSZ);
  cvt_f32_to_bf16<<<(H2 * H1 + 255) / 256, 256, 0, stream>>>(fc2_w, Wf2, H2 * H1);
  bn_fold_kernel<<<1, 64, 0, stream>>>(bn_g, bn_be, bn_mu, bn_var, conv_b, scale, shift);

  // Conv + BN + ReLU (dominant: ~65 GFLOP via bf16 WMMA, x read once via LDS)
  conv_bn_relu_kernel<<<BATCH, 128, 0, stream>>>(x, Wc, scale, shift, act);
  // FC1
  fc1_kernel<<<(BATCH / 16) * 4, 256, 0, stream>>>(act, Wf1, fc1_b, h1);
  // FC2 + parabit
  fc2_parabit_kernel<<<BATCH / 16, 256, 0, stream>>>(h1, Wf2, fc2_b, bit_w, bit_b, out);
}
